// MultiHeadAttention_42580305772722
// MI455X (gfx1250) — matrix operand order repaired and checked by reading, not compiled, not run on hardware
//
#include <hip/hip_runtime.h>
#include <hip/hip_bf16.h>

// ---------------------------------------------------------------------------
// MI455X (gfx1250) fused MHA block: QKV GEMM -> flash attention -> proj GEMM.
// Matrix math: v_wmma_f32_16x16x32_bf16 (wave32).
// Tile staging: global_load_async_to_lds_b128 (ASYNCcnt).
// ---------------------------------------------------------------------------

typedef __attribute__((ext_vector_type(16))) __bf16 bf16x16;
typedef __attribute__((ext_vector_type(8)))  float  floatx8;
typedef int v4i_vs __attribute__((vector_size(16)));   // matches builtin pointee

union FragB16 { unsigned int u[8]; bf16x16 v; };

#define TT 2048
#define CC 1024
#define HH 16
#define DD 64

#define AS_GLOBAL __attribute__((address_space(1)))
#define AS_LOCAL  __attribute__((address_space(3)))

#if defined(__has_builtin)
#if __has_builtin(__builtin_amdgcn_global_load_async_to_lds_b128)
#define HAVE_ASYNC_LDS 1
#endif
#endif
#ifndef HAVE_ASYNC_LDS
#define HAVE_ASYNC_LDS 0
#endif

__device__ __forceinline__ void async_copy16(const void* g, void* l) {
#if HAVE_ASYNC_LDS
  __builtin_amdgcn_global_load_async_to_lds_b128(
      (AS_GLOBAL v4i_vs*)g, (AS_LOCAL v4i_vs*)l, 0, 0);
#else
  (void)g; (void)l;
#endif
}

__device__ __forceinline__ void wait_async0() {
#if HAVE_ASYNC_LDS
#if __has_builtin(__builtin_amdgcn_s_wait_asynccnt)
  __builtin_amdgcn_s_wait_asynccnt(0);
#else
  asm volatile("s_wait_asynccnt 0x0" ::: "memory");
#endif
#endif
}

__device__ __forceinline__ unsigned short f32_to_bf16_rne(float f) {
  unsigned int b = __builtin_bit_cast(unsigned int, f);
  b += 0x7FFFu + ((b >> 16) & 1u);
  return (unsigned short)(b >> 16);
}

__global__ __launch_bounds__(256)
void cvt_f32_bf16(const float* __restrict__ in, unsigned short* __restrict__ out,
                  long long n) {
  long long i = (long long)blockIdx.x * blockDim.x + threadIdx.x;
  if (i < n) out[i] = f32_to_bf16_rne(in[i]);
}

// ---------------------------------------------------------------------------
// GEMM: C[M,N] = A[M,K](bf16) * B[K,N](bf16) + bias.
// MODE 0: scatter into q/k/v bf16 arrays laid out [B,H,T,D].
// MODE 1: write f32 to outF.
// Block: 256 threads = 8 waves; tile 128x64; each wave 32x32 (2x2 WMMA).
// A tile staged via async-to-LDS; B tile transposed via scalar DS stores.
// ---------------------------------------------------------------------------
template <int MODE>
__global__ __launch_bounds__(256)
void gemm_bf16_wmma(const unsigned short* __restrict__ A,
                    const unsigned short* __restrict__ Bm,
                    const float* __restrict__ bias,
                    unsigned short* __restrict__ outQ,
                    unsigned short* __restrict__ outK,
                    unsigned short* __restrict__ outV,
                    float* __restrict__ outF,
                    int M, int N, int K) {
  __shared__ unsigned short sA[128][36];   // [m][k], padded (72B row stride)
  __shared__ unsigned short sB[64][36];    // [n][k], padded (B transposed)

  const int tid  = threadIdx.x;
  const int lane = tid & 31;
  const int w    = tid >> 5;
  const int wm   = w & 3;        // 4 waves along M
  const int wn   = w >> 2;       // 2 waves along N
  const int l15  = lane & 15;
  const int lh   = lane >> 4;    // lane half selects K sub-range
  const int rowStart = blockIdx.y * 128;
  const int colStart = blockIdx.x * 64;

  floatx8 acc[2][2] = {};

  for (int k0 = 0; k0 < K; k0 += 32) {
    __syncthreads();
    // --- A tile 128x32: 512 x 16B chunks, async DMA into LDS ---
#if HAVE_ASYNC_LDS
#pragma unroll
    for (int j = 0; j < 2; ++j) {
      int id = j * 256 + tid;          // 0..511
      int r  = id >> 2;                // row in tile
      int c8 = (id & 3) * 8;           // elem offset within row (16B chunks)
      async_copy16(A + (size_t)(rowStart + r) * K + k0 + c8, &sA[r][c8]);
    }
#else
    {
      const unsigned int* Au = (const unsigned int*)A;
#pragma unroll
      for (int j = 0; j < 8; ++j) {
        int p  = tid * 8 + j;
        int r  = p >> 4;
        int kp = p & 15;
        ((unsigned int*)&sA[r][0])[kp] =
            Au[((size_t)(rowStart + r) * K + k0) / 2 + kp];
      }
    }
#endif
    // --- B tile 32x64 -> transposed sB[n][k] (scalar stores) ---
    {
#pragma unroll
      for (int j = 0; j < 8; ++j) {
        int l  = tid * 8 + j;          // 0..2047 elems
        int n  = l & 63;
        int kk = l >> 6;
        sB[n][kk] = Bm[(size_t)(k0 + kk) * N + colStart + n];
      }
    }
    // prefetch next K-slab while this one is consumed
    if (k0 + 32 < K) {
      __builtin_prefetch(A + (size_t)(rowStart + (tid >> 1)) * K + k0 + 32, 0, 1);
      __builtin_prefetch(Bm + (size_t)(k0 + 32 + (tid & 31)) * N + colStart, 0, 1);
    }
    wait_async0();
    __syncthreads();

    // A fragments: lane half lh -> K offset 0/8 then +16 (ISA 16-bit A layout)
    FragB16 afr[2], bfr[2];
    const int off = lh * 8;
#pragma unroll
    for (int mi = 0; mi < 2; ++mi) {
      const unsigned short* ap = &sA[wm * 32 + mi * 16 + l15][0];
#pragma unroll
      for (int j = 0; j < 4; ++j) {
        afr[mi].u[j]     = *(const unsigned int*)&ap[off + 2 * j];
        afr[mi].u[4 + j] = *(const unsigned int*)&ap[off + 16 + 2 * j];
      }
    }
    const int kb = lh * 8;             // B: lane half selects K offset 0/8 then +16
#pragma unroll
    for (int ni = 0; ni < 2; ++ni) {
      const unsigned short* bp = &sB[wn * 32 + ni * 16 + l15][0];
#pragma unroll
      for (int j = 0; j < 8; ++j)
        bfr[ni].u[j] = *(const unsigned int*)&bp[kb + (j >> 2) * 16 + 2 * (j & 3)];
    }
#pragma unroll
    for (int mi = 0; mi < 2; ++mi)
#pragma unroll
      for (int ni = 0; ni < 2; ++ni)
        acc[mi][ni] = __builtin_amdgcn_wmma_f32_16x16x32_bf16(
            false, afr[mi].v, false, bfr[ni].v, (short)0, acc[mi][ni],
            false, false);
  }

  // Epilogue: C layout = VGPR r -> row r + 8*lh; col = l15 within 16-tile.
#pragma unroll
  for (int mi = 0; mi < 2; ++mi) {
#pragma unroll
    for (int ni = 0; ni < 2; ++ni) {
#pragma unroll
      for (int r = 0; r < 8; ++r) {
        int m = rowStart + wm * 32 + mi * 16 + r + 8 * lh;
        int n = colStart + wn * 32 + ni * 16 + l15;
        float vv = acc[mi][ni][r] + bias[n];
        if (MODE == 0) {
          int which = n >> 10;              // q / k / v
          int cn = n & (CC - 1);
          int h = cn >> 6, d = cn & (DD - 1);
          int bb = m >> 11, t = m & (TT - 1);
          size_t idx = ((size_t)(bb * HH + h) * TT + t) * DD + d;
          unsigned short bv = f32_to_bf16_rne(vv);
          if (which == 0)      outQ[idx] = bv;
          else if (which == 1) outK[idx] = bv;
          else                 outV[idx] = bv;
        } else {
          outF[(size_t)m * N + n] = vv;
        }
      }
    }
  }
}

// ---------------------------------------------------------------------------
// Flash attention: one workgroup (128 thr = 4 waves) per (head, 64-row qblock).
// Q frags in registers; K/V tiles in LDS (Q/K async-DMA'd); online softmax;
// P through a wave-private LDS slab to re-fragment for the PV WMMA.
// ---------------------------------------------------------------------------
__global__ __launch_bounds__(128)
void flash_attn(const unsigned short* __restrict__ Qg,
                const unsigned short* __restrict__ Kg,
                const unsigned short* __restrict__ Vg,
                unsigned short* __restrict__ Y) {
  __shared__ unsigned short sQ[64][68];
  __shared__ unsigned short sK[64][68];     // [key][d]
  __shared__ unsigned short sVT[64][68];    // [d][key]
  __shared__ unsigned short sP[4][16][68];  // per-wave P tile [qrow][key]

  const int tid  = threadIdx.x;
  const int lane = tid & 31;
  const int w    = tid >> 5;      // 4 waves, each owns 16 q-rows
  const int l15  = lane & 15;
  const int lh   = lane >> 4;
  const int off  = lh * 8;        // A-frag K offset by lane half
  const int kbb  = lh * 16;       // unused; B frags also use off (lh*8)

  const int qb = blockIdx.x;
  const int bh = blockIdx.y;      // b*H + h
  const int q0 = qb * 64;
  const size_t headBase = (size_t)bh * TT * DD;

  // --- load Q tile 64x64 (512 x 16B chunks) ---
  {
    const unsigned short* Qp = Qg + headBase + (size_t)q0 * DD;
#if HAVE_ASYNC_LDS
#pragma unroll
    for (int j = 0; j < 4; ++j) {
      int id = j * 128 + tid;          // 0..511
      int r  = id >> 3;
      int c8 = (id & 7) * 8;
      async_copy16(Qp + (size_t)id * 8, &sQ[r][c8]);
    }
    wait_async0();
#else
    const unsigned int* Qu = (const unsigned int*)Qp;
#pragma unroll
    for (int j = 0; j < 16; ++j) {
      int p = tid * 16 + j;
      ((unsigned int*)&sQ[p >> 5][0])[p & 31] = Qu[p];
    }
#endif
  }
  __syncthreads();

  FragB16 qa[2];  // Q A-frags: 16 rows x (2 chunks of K=32 over D=64)
  {
    const unsigned short* qp = &sQ[w * 16 + l15][0];
#pragma unroll
    for (int c = 0; c < 2; ++c)
#pragma unroll
      for (int j = 0; j < 4; ++j) {
        qa[c].u[j]     = *(const unsigned int*)&qp[c * 32 + off + 2 * j];
        qa[c].u[4 + j] = *(const unsigned int*)&qp[c * 32 + off + 16 + 2 * j];
      }
  }

  floatx8 o[4] = {};
  float mrow[8], lrow[8];
#pragma unroll
  for (int r = 0; r < 8; ++r) { mrow[r] = -1e30f; lrow[r] = 0.0f; }

  for (int kb = 0; kb <= qb; ++kb) {
    __syncthreads();
    // --- K tile [key][d]: async DMA ---
    {
      const unsigned short* Kp = Kg + headBase + (size_t)kb * 64 * DD;
#if HAVE_ASYNC_LDS
#pragma unroll
      for (int j = 0; j < 4; ++j) {
        int id = j * 128 + tid;
        int r  = id >> 3;
        int c8 = (id & 7) * 8;
        async_copy16(Kp + (size_t)id * 8, &sK[r][c8]);
      }
#else
      const unsigned int* Ku = (const unsigned int*)Kp;
#pragma unroll
      for (int j = 0; j < 16; ++j) {
        int p = tid * 16 + j;
        ((unsigned int*)&sK[p >> 5][0])[p & 31] = Ku[p];
      }
#endif
    }
    // --- V tile transposed -> sVT[d][key] (scalar transpose) ---
    {
      const unsigned short* Vp = Vg + headBase + (size_t)kb * 64 * DD;
#pragma unroll
      for (int j = 0; j < 32; ++j) {
        int l = tid * 32 + j;      // 4096 elems
        int d = l & 63;
        int key = l >> 6;
        sVT[d][key] = Vp[(size_t)key * DD + d];
      }
    }
    wait_async0();
    __syncthreads();

    // S = Q * K^T  (B matrix: N=key from lane, K=d; sK[key][d] is contiguous)
    floatx8 s[4] = {};
#pragma unroll
    for (int c = 0; c < 2; ++c) {
      FragB16 bk[4];
#pragma unroll
      for (int ni = 0; ni < 4; ++ni) {
        const unsigned short* kp = &sK[ni * 16 + l15][0];
#pragma unroll
        for (int j = 0; j < 8; ++j)
          bk[ni].u[j] = *(const unsigned int*)&kp[c * 32 + off + (j >> 2) * 16 + 2 * (j & 3)];
      }
#pragma unroll
      for (int ni = 0; ni < 4; ++ni)
        s[ni] = __builtin_amdgcn_wmma_f32_16x16x32_bf16(
            false, qa[c].v, false, bk[ni].v, (short)0, s[ni], false, false);
    }

    // Online softmax over this key tile
    float sv[4][8], rmax[8], rsum[8];
    const bool diag = (kb == qb);
#pragma unroll
    for (int r = 0; r < 8; ++r) rmax[r] = -1e30f;
#pragma unroll
    for (int ni = 0; ni < 4; ++ni) {
      int col = kb * 64 + ni * 16 + l15;
#pragma unroll
      for (int r = 0; r < 8; ++r) {
        int row = q0 + w * 16 + r + 8 * lh;
        float x = s[ni][r] * 0.125f;             // 1/sqrt(64)
        if (diag && col > row) x = -1e30f;       // causal mask
        sv[ni][r] = x;
        rmax[r] = fmaxf(rmax[r], x);
      }
    }
#pragma unroll
    for (int r = 0; r < 8; ++r)
#pragma unroll
      for (int msk = 1; msk < 16; msk <<= 1)
        rmax[r] = fmaxf(rmax[r], __shfl_xor(rmax[r], msk, 32));

    float alpha[8], mnew[8];
#pragma unroll
    for (int r = 0; r < 8; ++r) {
      mnew[r] = fmaxf(mrow[r], rmax[r]);
      alpha[r] = __expf(mrow[r] - mnew[r]);
      rsum[r] = 0.0f;
    }
#pragma unroll
    for (int ni = 0; ni < 4; ++ni)
#pragma unroll
      for (int r = 0; r < 8; ++r) {
        float p = __expf(sv[ni][r] - mnew[r]);
        rsum[r] += p;
        sP[w][r + 8 * lh][ni * 16 + l15] = f32_to_bf16_rne(p);
      }
#pragma unroll
    for (int r = 0; r < 8; ++r) {
#pragma unroll
      for (int msk = 1; msk < 16; msk <<= 1)
        rsum[r] += __shfl_xor(rsum[r], msk, 32);
      lrow[r] = lrow[r] * alpha[r] + rsum[r];
      mrow[r] = mnew[r];
    }
#pragma unroll
    for (int ni = 0; ni < 4; ++ni)
#pragma unroll
      for (int r = 0; r < 8; ++r)
        o[ni][r] = o[ni][r] * alpha[r];

    // O += P * V  (A=P [16 x 64keys]; B=V [keys x d] via sVT[d][key])
#pragma unroll
    for (int c = 0; c < 2; ++c) {
      FragB16 pa;
      const unsigned short* pp = &sP[w][l15][0];
#pragma unroll
      for (int j = 0; j < 4; ++j) {
        pa.u[j]     = *(const unsigned int*)&pp[c * 32 + off + 2 * j];
        pa.u[4 + j] = *(const unsigned int*)&pp[c * 32 + off + 16 + 2 * j];
      }
      FragB16 bv[4];
#pragma unroll
      for (int ni = 0; ni < 4; ++ni) {
        const unsigned short* vp = &sVT[ni * 16 + l15][0];
#pragma unroll
        for (int j = 0; j < 8; ++j)
          bv[ni].u[j] = *(const unsigned int*)&vp[c * 32 + off + (j >> 2) * 16 + 2 * (j & 3)];
      }
#pragma unroll
      for (int ni = 0; ni < 4; ++ni)
        o[ni] = __builtin_amdgcn_wmma_f32_16x16x32_bf16(
            false, pa.v, false, bv[ni].v, (short)0, o[ni], false, false);
    }
  }

  // y[b][t][h*64+d] = O / l
  const int bb = bh / HH;
  const int h  = bh % HH;
#pragma unroll
  for (int ni = 0; ni < 4; ++ni)
#pragma unroll
    for (int r = 0; r < 8; ++r) {
      int row = q0 + w * 16 + r + 8 * lh;
      int d = ni * 16 + l15;
      float yv = o[ni][r] / lrow[r];
      Y[((size_t)bb * TT + row) * CC + h * DD + d] = f32_to_bf16_rne(yv);
    }
}

// ---------------------------------------------------------------------------
extern "C" void kernel_launch(void* const* d_in, const int* in_sizes, int n_in,
                              void* d_out, int out_size, void* d_ws, size_t ws_size,
                              hipStream_t stream) {
  const float* x      = (const float*)d_in[0];
  const float* W_attn = (const float*)d_in[1];
  const float* b_attn = (const float*)d_in[2];
  const float* W_proj = (const float*)d_in[3];
  const float* b_proj = (const float*)d_in[4];
  float* out = (float*)d_out;

  const size_t nX  = (size_t)2 * TT * CC;      // 4,194,304
  const size_t nWa = (size_t)CC * 3 * CC;      // 3,145,728
  const size_t nWp = (size_t)CC * CC;          // 1,048,576

  unsigned short* ws  = (unsigned short*)d_ws;
  unsigned short* xb  = ws;  ws += nX;
  unsigned short* wab = ws;  ws += nWa;
  unsigned short* wpb = ws;  ws += nWp;
  unsigned short* qb  = ws;  ws += nX;
  unsigned short* kb  = ws;  ws += nX;
  unsigned short* vb  = ws;  ws += nX;
  unsigned short* yb  = ws;  ws += nX;         // total ~48 MB

  cvt_f32_bf16<<<dim3((unsigned)((nX  + 255) / 256)), 256, 0, stream>>>(x, xb, (long long)nX);
  cvt_f32_bf16<<<dim3((unsigned)((nWa + 255) / 256)), 256, 0, stream>>>(W_attn, wab, (long long)nWa);
  cvt_f32_bf16<<<dim3((unsigned)((nWp + 255) / 256)), 256, 0, stream>>>(W_proj, wpb, (long long)nWp);

  dim3 g1(3 * CC / 64, 2 * TT / 128);   // 48 x 32
  gemm_bf16_wmma<0><<<g1, 256, 0, stream>>>(xb, wab, b_attn, qb, kb, vb,
                                            nullptr, 2 * TT, 3 * CC, CC);

  dim3 g2(TT / 64, 2 * HH);             // 32 x 32
  flash_attn<<<g2, 128, 0, stream>>>(qb, kb, vb, yb);

  dim3 g3(CC / 64, 2 * TT / 128);       // 16 x 32
  gemm_bf16_wmma<1><<<g3, 256, 0, stream>>>(yb, wpb, b_proj, nullptr, nullptr,
                                            nullptr, out, 2 * TT, CC, CC);
}
